// IPAGNNInterpolant_35270271434821
// MI455X (gfx1250) — compile-verified, matrix operand order checked by reading, not run on hardware
//
#include <hip/hip_runtime.h>
#include <hip/hip_bf16.h>
#include <math.h>

// Problem constants (from reference)
#define BB 16
#define NN 256
#define LL 4
#define HH 128
#define VV 512
#define OV 1000
#define H4 512   // 4*H

typedef __attribute__((ext_vector_type(16))) _Float16 v16h;
typedef __attribute__((ext_vector_type(8)))  _Float16 v8h;
typedef __attribute__((ext_vector_type(8)))  float    v8f;

// Fragment-swizzled weight block size per matrix: 32 colTiles * 4 kk * 32 lanes * 16
#define WBLK (32 * 4 * 32 * 16)   // 65536 halves = 128 KB per matrix

// ---------------- WMMA helpers ----------------

__device__ __forceinline__ v8f wmma32(v16h a, v16h b, v8f c) {
  // D = A(16x32 f16) * B(32x16 f16) + C(16x16 f32)
  return __builtin_amdgcn_wmma_f32_16x16x32_f16(
      false, a, false, b, (short)0, c, false, false);
}

union frag_u {
  v16h v;
  struct { v8h lo, hi; } s;
};

// A fragment: 16x32 tile from row-major f16 buffer (leading dim ld halves).
// Lane l: row m = l&15, half hl = l>>4.  Holds K=[hl*8,hl*8+8) then
// K=[16+hl*8,16+hl*8+8) -> two contiguous 16-byte runs => 2x b128 loads.
__device__ __forceinline__ v16h load_a(const _Float16* __restrict__ p, int ld) {
  const int lane = threadIdx.x & 31;
  const int m  = lane & 15;
  const int hl = lane >> 4;
  const _Float16* row = p + m * ld + hl * 8;
  frag_u u;
  u.s.lo = *(const v8h*)(row);
  u.s.hi = *(const v8h*)(row + 16);
  return u.v;
}

// B fragment from pre-swizzled weights: each lane's 16 halves are contiguous.
// Layout: mat[((colTile*4 + kk)*32 + lane)*16 + elem]
__device__ __forceinline__ v16h load_b_sw(const _Float16* __restrict__ mat,
                                          int colTile, int kk) {
  const int lane = threadIdx.x & 31;
  const _Float16* p = mat + (((colTile * 4 + kk) * 32 + lane) << 4);
  frag_u u;
  u.s.lo = *(const v8h*)(p);
  u.s.hi = *(const v8h*)(p + 8);
  return u.v;
}

// CDNA5 hardware transcendental: v_tanh_f32 (TRANS pipe, gfx1250).
__device__ __forceinline__ float tanh_hw(float x) {
  float r;
  asm("v_tanh_f32 %0, %1" : "=v"(r) : "v"(x));
  return r;
}
// sigmoid(x) = 0.5*tanh(x/2) + 0.5  -> one v_tanh_f32 + FMAs
__device__ __forceinline__ float sigf(float x) {
  return __builtin_fmaf(tanh_hw(0.5f * x), 0.5f, 0.5f);
}

// One LSTM layer for this wave's 32 rows (2 tiles of 16). In-place c update;
// h double-buffered through per-wave LDS scratch (16x128).
__device__ __forceinline__ void lstm_layer(
    const _Float16* __restrict__ x, int x_ld,     // A input (rows of this wave)
    _Float16* h, _Float16* c,                     // state [NN][HH] in LDS
    const _Float16* __restrict__ wi,              // swizzled [32ct][4kk][32lane][16]
    const _Float16* __restrict__ wh,              // swizzled
    const float* __restrict__ bias,               // [4H]
    _Float16* hscr,                               // per-wave scratch [16][HH]
    int rowBase) {
  const int lane = threadIdx.x & 31;
  const int n  = lane & 15;
  const int mb = (lane >> 4) * 8;   // C/D layout: VGPR r holds row mb + r

#pragma unroll
  for (int tile = 0; tile < 2; ++tile) {
    const int rt = rowBase + tile * 16;
    for (int cg = 0; cg < 8; ++cg) {     // 8 column groups of 16 within H
      v8f acc0 = {}, acc1 = {}, acc2 = {}, acc3 = {};
#pragma unroll
      for (int kk = 0; kk < 4; ++kk) {   // K = 128 in chunks of 32
        v16h ax = load_a(x + rt * x_ld + kk * 32, x_ld);
        v16h ah = load_a(h + rt * HH  + kk * 32, HH);
        // colTile within the 512-wide gate matrix: gate g block at g*8 + cg
        acc0 = wmma32(ax, load_b_sw(wi, 0 * 8 + cg, kk), acc0);
        acc0 = wmma32(ah, load_b_sw(wh, 0 * 8 + cg, kk), acc0);
        acc1 = wmma32(ax, load_b_sw(wi, 1 * 8 + cg, kk), acc1);
        acc1 = wmma32(ah, load_b_sw(wh, 1 * 8 + cg, kk), acc1);
        acc2 = wmma32(ax, load_b_sw(wi, 2 * 8 + cg, kk), acc2);
        acc2 = wmma32(ah, load_b_sw(wh, 2 * 8 + cg, kk), acc2);
        acc3 = wmma32(ax, load_b_sw(wi, 3 * 8 + cg, kk), acc3);
        acc3 = wmma32(ah, load_b_sw(wh, 3 * 8 + cg, kk), acc3);
      }
      const int col = cg * 16 + n;
#pragma unroll
      for (int r = 0; r < 8; ++r) {
        const int m = rt + mb + r;
        const float zi = acc0[r] + bias[0 * HH + col];
        const float zf = acc1[r] + bias[1 * HH + col];
        const float zg = acc2[r] + bias[2 * HH + col];
        const float zo = acc3[r] + bias[3 * HH + col];
        const float ig = sigf(zi), fg = sigf(zf);
        const float gg = tanh_hw(zg), og = sigf(zo);
        const float cold = (float)c[m * HH + col];
        const float cn = __builtin_fmaf(fg, cold, ig * gg);
        c[m * HH + col] = (_Float16)cn;
        hscr[(mb + r) * HH + col] = (_Float16)(og * tanh_hw(cn));
      }
    }
    // commit new h tile (wave-private rows; no cross-wave sync needed)
    for (int idx = lane; idx < 16 * HH; idx += 32)
      h[(rt + (idx >> 7)) * HH + (idx & (HH - 1))] = hscr[idx];
  }
}

// ---------------- helper kernels ----------------

// Convert f32 weights into fragment-swizzled f16 layout:
//   dst[mat][colTile(32)][kk(4)][lane(32)][elem(16)]
__global__ void convert_weights(const float* __restrict__ wi0,
                                const float* __restrict__ wh0,
                                const float* __restrict__ wi1,
                                const float* __restrict__ wh1,
                                _Float16* __restrict__ w16) {
  const int t = blockIdx.x * blockDim.x + threadIdx.x;   // [0, 4*32*4*32)
  if (t >= 4 * 32 * 4 * 32) return;
  const int lane = t & 31;
  const int kk   = (t >> 5) & 3;
  const int ct   = (t >> 7) & 31;
  const int mat  = t >> 12;
  const float* W = (mat == 0) ? wi0 : (mat == 1) ? wh0 : (mat == 2) ? wi1 : wh1;
  const int hl = lane >> 4;
  const int n  = lane & 15;
  _Float16* dst = w16 + (size_t)t * 16;
#pragma unroll
  for (int v = 0; v < 8; ++v) {
#pragma unroll
    for (int e = 0; e < 2; ++e) {
      const int kl = ((v >> 2) << 4) + hl * 8 + ((v & 3) << 1) + e;
      const int k = kk * 32 + kl;
      dst[v * 2 + e] = (_Float16)W[k * H4 + ct * 16 + n];
    }
  }
}

__global__ void embed_gather(const int* __restrict__ data,
                             const float* __restrict__ embed_w,
                             _Float16* __restrict__ emb16) {
  const long long i = (long long)blockIdx.x * blockDim.x + threadIdx.x;
  if (i >= (long long)BB * LL * NN * HH) return;
  const int h = (int)(i & (HH - 1));
  const int n = (int)((i >> 7) & (NN - 1));
  const int l = (int)((i >> 15) & (LL - 1));
  const int b = (int)(i >> 17);
  const int v = data[(b * NN + n) * LL + l];
  emb16[i] = (_Float16)embed_w[v * HH + h];
}

// CSR of incoming branch edges per example (deterministic serial build).
// Entry = src | (1<<31 if true-branch edge).
__global__ void build_csr(const int* __restrict__ tidx,
                          const int* __restrict__ fidx,
                          int* __restrict__ csr_off,
                          int* __restrict__ csr_src) {
  const int b = blockIdx.x;
  if (threadIdx.x != 0) return;
  const int* ti = tidx + b * NN;
  const int* fi = fidx + b * NN;
  int* off = csr_off + b * (NN + 1);
  int* sv  = csr_src + b * (2 * NN);
  int cnt[NN];
  for (int j = 0; j < NN; ++j) cnt[j] = 0;
  for (int i = 0; i < NN; ++i) { cnt[ti[i]]++; cnt[fi[i]]++; }
  int run = 0;
  for (int j = 0; j < NN; ++j) { off[j] = run; run += cnt[j]; cnt[j] = off[j]; }
  off[NN] = run;
  for (int i = 0; i < NN; ++i) {
    sv[cnt[ti[i]]++] = i | (int)0x80000000;
    sv[cnt[fi[i]]++] = i;
  }
}

// ---------------- main kernel: one workgroup per example ----------------

__global__ __launch_bounds__(256)
void ipagnn_main(const int* __restrict__ startI, const int* __restrict__ exitI,
                 const int* __restrict__ stepsI,
                 const float* __restrict__ b0, const float* __restrict__ b1,
                 const float* __restrict__ branch_w, const float* __restrict__ branch_b,
                 const float* __restrict__ out_w, const float* __restrict__ out_b,
                 const _Float16* __restrict__ w16,
                 const _Float16* __restrict__ emb16,
                 const int* __restrict__ csr_off, const int* __restrict__ csr_src,
                 _Float16* __restrict__ hsnew,  // [B][4][N][H] bounce buffer
                 float* __restrict__ outp) {
  // 293 KB of the 320 KB/WGP LDS: full recurrent state stays on-chip for all steps.
  __shared__ _Float16 s_c0[NN * HH];
  __shared__ _Float16 s_h0[NN * HH];
  __shared__ _Float16 s_c1[NN * HH];
  __shared__ _Float16 s_h1[NN * HH];
  __shared__ _Float16 s_scr[8 * 16 * HH];   // per-wave h double-buffer
  __shared__ _Float16 s_exit[4 * HH];
  __shared__ float s_ip[NN], s_ipn[NN], s_wt[NN], s_wf[NN];

  const int b    = blockIdx.x;
  const int tid  = threadIdx.x;
  const int wave = tid >> 5;
  const int rowBase = wave * 32;
  const int xi = exitI[b];
  const int nsteps = stepsI[b];

  const _Float16* Wi0 = w16 + 0 * WBLK;
  const _Float16* Wh0 = w16 + 1 * WBLK;
  const _Float16* Wi1 = w16 + 2 * WBLK;
  const _Float16* Wh1 = w16 + 3 * WBLK;
  const int* off  = csr_off + b * (NN + 1);
  const int* esrc = csr_src + b * (2 * NN);
  _Float16* hsn = hsnew + (long long)b * 4 * NN * HH;
  _Float16* scr = s_scr + wave * 16 * HH;

  // Pull the f16 weights toward L2 (global_prefetch_b8 path).
  for (long long o = (long long)tid * 128; o < (long long)4 * WBLK * 2; o += 256 * 128)
    __builtin_prefetch((const char*)w16 + o, 0, 0);

  for (int i = tid; i < NN * HH; i += 256) {
    s_c0[i] = (_Float16)0.f; s_h0[i] = (_Float16)0.f;
    s_c1[i] = (_Float16)0.f; s_h1[i] = (_Float16)0.f;
  }
  if (tid < NN) s_ip[tid] = (tid == startI[b]) ? 1.0f : 0.0f;
  __syncthreads();

  for (int step = 0; step < nsteps; ++step) {
    // snapshot hs[exit] (contrib override uses pre-execute values)
    for (int t = tid; t < HH; t += 256) {
      s_exit[0 * HH + t] = s_c0[xi * HH + t];
      s_exit[1 * HH + t] = s_h0[xi * HH + t];
      s_exit[2 * HH + t] = s_c1[xi * HH + t];
      s_exit[3 * HH + t] = s_h1[xi * HH + t];
    }
    __syncthreads();

    // execute(): 4 tokens x 2 LSTM layers; rows are wave-private -> no barriers
    for (int t = 0; t < LL; ++t) {
      const _Float16* x = emb16 + ((long long)(b * LL + t) * NN) * HH;
      lstm_layer(x,    HH, s_h0, s_c0, Wi0, Wh0, b0, scr, rowBase);
      lstm_layer(s_h0, HH, s_h1, s_c1, Wi1, Wh1, b1, scr, rowBase);
    }
    __syncthreads();

    // contrib.at[exit] = old hs[exit]
    for (int t = tid; t < HH; t += 256) {
      s_c0[xi * HH + t] = s_exit[0 * HH + t];
      s_h0[xi * HH + t] = s_exit[1 * HH + t];
      s_c1[xi * HH + t] = s_exit[2 * HH + t];
      s_h1[xi * HH + t] = s_exit[3 * HH + t];
    }
    __syncthreads();

    // branch head: logits = concat @ (512x2) + b, softmax, weight by ip
    {
      const int i = tid;
      float a0 = branch_b[0], a1 = branch_b[1];
      for (int k = 0; k < HH; ++k) {
        float v;
        v = (float)s_c0[i * HH + k]; a0 += v * branch_w[(0 * HH + k) * 2]; a1 += v * branch_w[(0 * HH + k) * 2 + 1];
        v = (float)s_h0[i * HH + k]; a0 += v * branch_w[(1 * HH + k) * 2]; a1 += v * branch_w[(1 * HH + k) * 2 + 1];
        v = (float)s_c1[i * HH + k]; a0 += v * branch_w[(2 * HH + k) * 2]; a1 += v * branch_w[(2 * HH + k) * 2 + 1];
        v = (float)s_h1[i * HH + k]; a0 += v * branch_w[(3 * HH + k) * 2]; a1 += v * branch_w[(3 * HH + k) * 2 + 1];
      }
      const float mx = fmaxf(a0, a1);
      const float e0 = __expf(a0 - mx), e1 = __expf(a1 - mx);
      const float inv = __builtin_amdgcn_rcpf(e0 + e1);
      const float ipv = s_ip[i];
      s_wt[i] = e0 * inv * ipv;
      s_wf[i] = e1 * inv * ipv;
    }
    __syncthreads();

    // gather-style soft aggregation via incoming-edge CSR (deterministic)
    {
      const int j = tid;
      const int eb = off[j], ee = off[j + 1];
      float ipn = 0.0f;
      for (int e = eb; e < ee; ++e) {
        const int pk = esrc[e];
        ipn += (pk < 0) ? s_wt[pk & 0x3FF] : s_wf[pk & 0x3FF];
      }
      s_ipn[j] = ipn;
      const float inv = __builtin_amdgcn_rcpf(ipn + 1e-7f);
      for (int col = 0; col < HH; ++col) {
        float a0 = 0.f, a1 = 0.f, a2 = 0.f, a3 = 0.f;
        for (int e = eb; e < ee; ++e) {
          const int pk = esrc[e];
          const int src = pk & 0x3FF;
          const float w = (pk < 0) ? s_wt[src] : s_wf[src];
          a0 += w * (float)s_c0[src * HH + col];
          a1 += w * (float)s_h0[src * HH + col];
          a2 += w * (float)s_c1[src * HH + col];
          a3 += w * (float)s_h1[src * HH + col];
        }
        hsn[(0 * NN + j) * HH + col] = (_Float16)(a0 * inv);
        hsn[(1 * NN + j) * HH + col] = (_Float16)(a1 * inv);
        hsn[(2 * NN + j) * HH + col] = (_Float16)(a2 * inv);
        hsn[(3 * NN + j) * HH + col] = (_Float16)(a3 * inv);
      }
    }
    __syncthreads();

    // commit new state back into LDS
    {
      const int j = tid;
      s_ip[j] = s_ipn[j];
      for (int col = 0; col < HH; ++col) {
        s_c0[j * HH + col] = hsn[(0 * NN + j) * HH + col];
        s_h0[j * HH + col] = hsn[(1 * NN + j) * HH + col];
        s_c1[j * HH + col] = hsn[(2 * NN + j) * HH + col];
        s_h1[j * HH + col] = hsn[(3 * NN + j) * HH + col];
      }
    }
    __syncthreads();
  }

  // readout: h1[exit] @ out_w + out_b
  for (int o = tid; o < OV; o += 256) {
    float acc = out_b[o];
    for (int k = 0; k < HH; ++k)
      acc += (float)s_h1[xi * HH + k] * out_w[k * OV + o];
    outp[b * OV + o] = acc;
  }
}

// ---------------- launch ----------------

extern "C" void kernel_launch(void* const* d_in, const int* in_sizes, int n_in,
                              void* d_out, int out_size, void* d_ws, size_t ws_size,
                              hipStream_t stream) {
  const int*   data   = (const int*)  d_in[0];
  const int*   tidx   = (const int*)  d_in[1];
  const int*   fidx   = (const int*)  d_in[2];
  const int*   startI = (const int*)  d_in[3];
  const int*   exitI  = (const int*)  d_in[4];
  const int*   stepsI = (const int*)  d_in[5];
  const float* embedw = (const float*)d_in[6];
  const float* Wi0    = (const float*)d_in[7];
  const float* Wh0    = (const float*)d_in[8];
  const float* b0     = (const float*)d_in[9];
  const float* Wi1    = (const float*)d_in[10];
  const float* Wh1    = (const float*)d_in[11];
  const float* b1     = (const float*)d_in[12];
  const float* brw    = (const float*)d_in[13];
  const float* brb    = (const float*)d_in[14];
  const float* outw   = (const float*)d_in[15];
  const float* outb   = (const float*)d_in[16];
  float* outp = (float*)d_out;

  // Workspace layout
  char* ws = (char*)d_ws;
  _Float16* w16   = (_Float16*)(ws);                                   // 512 KB
  _Float16* emb16 = (_Float16*)(ws + 524288);                          // 4 MB
  _Float16* hsn   = (_Float16*)(ws + 524288 + 4194304);                // 4 MB
  int* csr_off    = (int*)(ws + 524288 + 4194304 + 4194304);           // 16.4 KB
  int* csr_src    = (int*)(ws + 524288 + 4194304 + 4194304 + BB * (NN + 1) * 4);
  (void)ws_size; (void)in_sizes; (void)n_in; (void)out_size;

  convert_weights<<<(4 * 32 * 4 * 32 + 255) / 256, 256, 0, stream>>>(Wi0, Wh0, Wi1, Wh1, w16);
  embed_gather<<<(BB * LL * NN * HH + 255) / 256, 256, 0, stream>>>(data, embedw, emb16);
  build_csr<<<BB, 1, 0, stream>>>(tidx, fidx, csr_off, csr_src);
  ipagnn_main<<<BB, 256, 0, stream>>>(startI, exitI, stepsI, b0, b1, brw, brb,
                                      outw, outb, w16, emb16, csr_off, csr_src,
                                      hsn, outp);
}
